// MLPPredictor_36541581754705
// MI455X (gfx1250) — compile-verified, matrix operand order checked by reading, not run on hardware
//
#include <hip/hip_runtime.h>
#include <hip/hip_bf16.h>

typedef __attribute__((ext_vector_type(16))) _Float16 v16h;
typedef __attribute__((ext_vector_type(4)))  _Float16 v4h;
typedef __attribute__((ext_vector_type(8)))  float    v8f;

#define HF 128            // feature dim
#define NOUT 256          // U|V concatenated output dim
#define BFRAG_BYTES 65536 // 16 jt * 4 ks * 32 lanes * 32B

// ---------------------------------------------------------------------------
// Kernel 0: pre-swizzle W1 (f32 [128][256], torch out-x-in) into per-lane WMMA
// B-fragments (f16): one v16h per (jt, ks, lane).
// Logical B (K=128 x N=256): n<128: B[k][n]=W1[n][k]; n>=128: B[k][n]=W1[n-128][128+k]
// 16-bit B 32x16 fragment layout (lane = N, lanes 16-31 take K+8 / K+24):
//   elem e<8 : K = 32*ks + klo + e        klo = (lane<16 ? 0 : 8)
//   elem e>=8: K = 32*ks + klo + 16 + e-8
// ---------------------------------------------------------------------------
__global__ void pack_w1_kernel(const float* __restrict__ W1, v16h* __restrict__ bfrag) {
    int t = blockIdx.x * blockDim.x + threadIdx.x;   // 0..2047
    if (t >= 16 * 4 * 32) return;
    int jt   = t >> 7;
    int ks   = (t >> 5) & 3;
    int lane = t & 31;
    int n    = jt * 16 + (lane & 15);
    int klo  = (lane & 16) ? 8 : 0;
    v16h f;
#pragma unroll
    for (int e = 0; e < 16; ++e) {
        int k = ks * 32 + klo + ((e < 8) ? e : (16 + e - 8));
        float val = (n < HF) ? W1[n * NOUT + k]
                             : W1[(n - HF) * NOUT + HF + k];
        f[e] = (_Float16)val;
    }
    bfrag[t] = f;
}

// ---------------------------------------------------------------------------
// Kernel 1: C[n][0:256] = h[n] @ B via v_wmma_f32_16x16x32_f16, stored f16.
// 256 threads = 8 waves; block covers 128 rows. A fragments built once per
// wave, reused over all 16 N-tiles. B fragments double-buffered (prefetch
// next tile before current tile's WMMAs -> loads overlap matrix ops).
// C is row-padded to gridDim.x*128 rows so stores are UNGUARDED; A-row is
// clamped (padding rows hold garbage that is never read).
// ---------------------------------------------------------------------------
__global__ void gemm_uv_kernel(const float* __restrict__ h,
                               const v16h* __restrict__ bfrag,
                               _Float16* __restrict__ C, int nNodes) {
    int wave = threadIdx.x >> 5;
    int lane = threadIdx.x & 31;
    int rowA = blockIdx.x * 128 + wave * 16 + (lane & 15);
    int rowL = rowA < nNodes ? rowA : nNodes - 1;   // clamp: always a valid load
    int klo  = (lane & 16) ? 8 : 0;
    const float* hrow = h + (size_t)rowL * HF;

    // Build A fragments for 4 K-steps (16x32 f16 layout, M = lane%16)
    v16h a[4];
#pragma unroll
    for (int s = 0; s < 4; ++s) {
        const float* p = hrow + s * 32 + klo;
        float4 f0 = *(const float4*)(p);
        float4 f1 = *(const float4*)(p + 4);
        float4 f2 = *(const float4*)(p + 16);
        float4 f3 = *(const float4*)(p + 20);
        a[s][0]  = (_Float16)f0.x; a[s][1]  = (_Float16)f0.y;
        a[s][2]  = (_Float16)f0.z; a[s][3]  = (_Float16)f0.w;
        a[s][4]  = (_Float16)f1.x; a[s][5]  = (_Float16)f1.y;
        a[s][6]  = (_Float16)f1.z; a[s][7]  = (_Float16)f1.w;
        a[s][8]  = (_Float16)f2.x; a[s][9]  = (_Float16)f2.y;
        a[s][10] = (_Float16)f2.z; a[s][11] = (_Float16)f2.w;
        a[s][12] = (_Float16)f3.x; a[s][13] = (_Float16)f3.y;
        a[s][14] = (_Float16)f3.z; a[s][15] = (_Float16)f3.w;
    }

    int col_lo = lane & 15;
    size_t rbase = (size_t)(blockIdx.x * 128 + wave * 16 + ((lane & 16) ? 8 : 0));

    // Prefetch B fragments for jt=0
    v16h bnxt[4];
#pragma unroll
    for (int s = 0; s < 4; ++s) bnxt[s] = bfrag[s * 32 + lane];

#pragma unroll
    for (int jt = 0; jt < 16; ++jt) {
        v16h bc0 = bnxt[0], bc1 = bnxt[1], bc2 = bnxt[2], bc3 = bnxt[3];
        if (jt < 15) {
#pragma unroll
            for (int s = 0; s < 4; ++s)
                bnxt[s] = bfrag[((jt + 1) * 4 + s) * 32 + lane];
        }
        v8f acc = {};
        acc = __builtin_amdgcn_wmma_f32_16x16x32_f16(false, a[0], false, bc0, (short)0, acc, false, false);
        acc = __builtin_amdgcn_wmma_f32_16x16x32_f16(false, a[1], false, bc1, (short)0, acc, false, false);
        acc = __builtin_amdgcn_wmma_f32_16x16x32_f16(false, a[2], false, bc2, (short)0, acc, false, false);
        acc = __builtin_amdgcn_wmma_f32_16x16x32_f16(false, a[3], false, bc3, (short)0, acc, false, false);

        // Convert all 8 results into distinct temps FIRST, then store
        // back-to-back (avoids per-store s_wait_xcnt on a reused data VGPR).
        _Float16 t0 = (_Float16)acc[0];
        _Float16 t1 = (_Float16)acc[1];
        _Float16 t2 = (_Float16)acc[2];
        _Float16 t3 = (_Float16)acc[3];
        _Float16 t4 = (_Float16)acc[4];
        _Float16 t5 = (_Float16)acc[5];
        _Float16 t6 = (_Float16)acc[6];
        _Float16 t7 = (_Float16)acc[7];

        int col = jt * 16 + col_lo;
        _Float16* cp = C + rbase * NOUT + col;
        cp[0 * NOUT] = t0;
        cp[1 * NOUT] = t1;
        cp[2 * NOUT] = t2;
        cp[3 * NOUT] = t3;
        cp[4 * NOUT] = t4;
        cp[5 * NOUT] = t5;
        cp[6 * NOUT] = t6;
        cp[7 * NOUT] = t7;
    }
}

// ---------------------------------------------------------------------------
// Kernel 2: wave processes 32 edges. Indices loaded once (coalesced, lane l
// owns edge e0+l) and broadcast via __shfl. Per edge: lanes gather 4 f16
// features of U[src] / V[dst] (coalesced 256B rows, L2-resident), ReLU+dot in
// f32. Cross-lane reduce via conflict-free 32x33 LDS transpose; coalesced
// 32-wide sigmoid+store.
// ---------------------------------------------------------------------------
__global__ void edge_mlp_kernel(const _Float16* __restrict__ C,
                                const long long* __restrict__ src,
                                const long long* __restrict__ dst,
                                const float* __restrict__ b1,
                                const float* __restrict__ W2,
                                const float* __restrict__ b2,
                                float* __restrict__ out, int nEdges) {
    __shared__ float red[8][32 * 33];
    int wave = threadIdx.x >> 5;
    int lane = threadIdx.x & 31;
    int e0 = (blockIdx.x * (blockDim.x >> 5) + wave) * 32;
    if (e0 >= nEdges) return;

    float4 bb = ((const float4*)b1)[lane];
    float4 ww = ((const float4*)W2)[lane];
    float bias2 = b2[0];

    int myE = e0 + lane;
    int sI = 0, dI = 0;
    if (myE < nEdges) { sI = (int)src[myE]; dI = (int)dst[myE]; }
    int cnt = nEdges - e0; if (cnt > 32) cnt = 32;

    float* rr = red[wave];

    // software-pipelined gather: prefetch edge i+1 while computing edge i
    int s0 = __shfl(sI, 0, 32);
    int d0 = __shfl(dI, 0, 32);
    v4h u = *(const v4h*)(C + (size_t)s0 * NOUT + lane * 4);
    v4h v = *(const v4h*)(C + (size_t)d0 * NOUT + HF + lane * 4);

    for (int i = 0; i < cnt; ++i) {
        v4h uc = u, vc = v;
        if (i + 1 < cnt) {
            int sn = __shfl(sI, i + 1, 32);
            int dn = __shfl(dI, i + 1, 32);
            u = *(const v4h*)(C + (size_t)sn * NOUT + lane * 4);
            v = *(const v4h*)(C + (size_t)dn * NOUT + HF + lane * 4);
        }
        float h0 = fmaxf((float)uc[0] + (float)vc[0] + bb.x, 0.f);
        float h1 = fmaxf((float)uc[1] + (float)vc[1] + bb.y, 0.f);
        float h2 = fmaxf((float)uc[2] + (float)vc[2] + bb.z, 0.f);
        float h3 = fmaxf((float)uc[3] + (float)vc[3] + bb.w, 0.f);
        rr[i * 33 + lane] = h0 * ww.x + h1 * ww.y + h2 * ww.z + h3 * ww.w;
    }

    // lane l sums the 32 partials of edge e0+l (bank-conflict-free: stride 33)
    float sum = 0.f;
    const float* row = rr + lane * 33;
#pragma unroll
    for (int j = 0; j < 32; ++j) sum += row[j];

    if (myE < nEdges)
        out[myE] = 1.f / (1.f + expf(-(sum + bias2)));
}

extern "C" void kernel_launch(void* const* d_in, const int* in_sizes, int n_in,
                              void* d_out, int out_size, void* d_ws, size_t ws_size,
                              hipStream_t stream) {
    const float*     h   = (const float*)d_in[0];
    const long long* src = (const long long*)d_in[1];
    const long long* dst = (const long long*)d_in[2];
    const float*     W1  = (const float*)d_in[3];
    const float*     b1  = (const float*)d_in[4];
    const float*     W2  = (const float*)d_in[5];
    const float*     b2  = (const float*)d_in[6];
    float* out = (float*)d_out;

    int nNodes = in_sizes[0] / HF;
    int nEdges = in_sizes[1];

    v16h*     bfrag = (v16h*)d_ws;
    _Float16* C     = (_Float16*)((char*)d_ws + BFRAG_BYTES);

    // 1) Pre-swizzle W1 into WMMA B-fragments
    pack_w1_kernel<<<8, 256, 0, stream>>>(W1, bfrag);

    // 2) Node-level GEMM: C = h @ [W1a^T | W1b^T], f16 output (row-padded)
    int gemmBlocks = (nNodes + 127) / 128;
    gemm_uv_kernel<<<gemmBlocks, 256, 0, stream>>>(h, bfrag, C, nNodes);

    // 3) Edge stage: 32 edges per wave, 8 waves per block
    int edgeWaves  = (nEdges + 31) / 32;
    int edgeBlocks = (edgeWaves + 7) / 8;
    edge_mlp_kernel<<<edgeBlocks, 256, 0, stream>>>(C, src, dst, b1, W2, b2, out, nEdges);
}
